// FastAttention_34969623724566
// MI455X (gfx1250) — matrix operand order repaired and checked by reading, not compiled, not run on hardware
//
#include <hip/hip_runtime.h>
#include <hip/hip_bf16.h>

// ---------------------------------------------------------------------------
// FAVOR+ (Performer) fast attention for MI455X (gfx1250, wave32, WMMA).
// b=2 h=8 n=4096 d=64 m=256, CHUNK=128.
// Heavy GEMMs use v_wmma_f32_16x16x32_bf16 (bf16 operands, f32 accum).
// phi projection GEMM uses v_wmma_f32_16x16x4_f32 (guarded, scalar fallback).
// Per-chunk qb/kb staging uses the Tensor Data Mover (tensor_load_to_lds +
// s_wait_tensorcnt); next-chunk tiles are warmed with global_prefetch_b8.
// Scan state S[256x64] f32 lives in LDS for the whole (b,h) scan; per-chunk
// working set (qb/kb/vb bf16 + scores bf16 + S f32) = ~247KB LDS, which the
// CDNA5 WGP's 320KB LDS can hold.
// ---------------------------------------------------------------------------

typedef __attribute__((ext_vector_type(16))) __bf16 v16bf;
typedef __attribute__((ext_vector_type(8)))  float  v8f;
typedef __attribute__((ext_vector_type(2)))  float  v2f;
typedef __attribute__((ext_vector_type(4)))  unsigned v4u;
typedef __attribute__((ext_vector_type(4)))  int    v4i;
typedef __attribute__((ext_vector_type(8)))  int    v8i_;

#define B_      2
#define H_      8
#define N_      4096
#define D_      64
#define M_      256
#define CHUNK_  128
#define NCHUNK  (N_ / CHUNK_)
#define BH_     (B_ * H_)
#define NRM_    0.35355339059327373f   /* 64^-0.25 */
#define RATIO_  0.0625f                /* 256^-0.5 */
#define EPSK_   1e-4f
#define EPSD_   1e-6f

// ---- order-preserving float<->uint encoding for atomic max ----------------
__device__ __forceinline__ unsigned f32_ordered(float f) {
  unsigned u = __float_as_uint(f);
  return (u & 0x80000000u) ? ~u : (u | 0x80000000u);
}
__device__ __forceinline__ float ordered_f32(unsigned e) {
  unsigned u = (e & 0x80000000u) ? (e & 0x7FFFFFFFu) : ~e;
  return __uint_as_float(u);
}

// ---- Tensor Data Mover: 1-D contiguous global -> LDS copy -----------------
// D# packing per CDNA5 ISA (group0 128b, group1 256b).
// bytes must be a multiple of 8 and <= 512KB (tile_dim0 is 16-bit).
#if __has_builtin(__builtin_amdgcn_tensor_load_to_lds)
#define HAVE_TDM_ 1
__device__ __forceinline__ void tdm_copy_to_lds(unsigned lds_off,
                                                const void* gsrc,
                                                unsigned bytes) {
  unsigned long long ga = (unsigned long long)gsrc;
  unsigned n8 = bytes >> 3;                     // elements of 8 bytes
  v4u g0;
  g0[0] = 1u;                                   // count=1, user mode
  g0[1] = lds_off;                              // lds_addr (bytes)
  g0[2] = (unsigned)ga;                         // global_addr[31:0]
  g0[3] = (unsigned)(ga >> 32) | (2u << 30);    // global_addr[56:32] | type=2
  v8i_ g1;
  g1[0] = (int)(3u << 16);                      // wg_mask=0, data_size=3 (8B)
  g1[1] = (int)((n8 & 0xFFFFu) << 16);          // tensor_dim0[15:0]
  g1[2] = (int)(((n8 >> 16) & 0xFFFFu) | (1u << 16)); // td0[31:16], td1=1
  g1[3] = (int)((n8 & 0xFFFFu) << 16);          // td1[31:16]=0, tile_dim0=n8
  g1[4] = 0;                                    // tile_dim1=0, tile_dim2=0
  g1[5] = (int)n8;                              // tensor_dim0_stride[31:0]
  g1[6] = 0;                                    // td0_stride[47:32], td1_stride
  g1[7] = 0;
  v4i z4 = {};
#if defined(__clang_major__) && __clang_major__ >= 23
  v8i_ z8 = {};
  __builtin_amdgcn_tensor_load_to_lds(g0, g1, z4, z4, z8, 0);
#else
  __builtin_amdgcn_tensor_load_to_lds(g0, g1, z4, z4, 0);
#endif
}
#else
#define HAVE_TDM_ 0
#endif

// ---- bf16 WMMA fragment builders (contiguous k = 16h + s, same order in A and B)
// A 16x32 bf16: lane half h, M = lane&15.  VGPR v: K = 16h + 2v
__device__ __forceinline__ v16bf frag_a_bf16(const __bf16* A, int lda, int row0,
                                             int k0, int l, int h) {
  v16bf f;
  const __bf16* ar = A + (row0 + l) * lda + k0;
#pragma unroll
  for (int vv = 0; vv < 8; ++vv) {
    int kk = 16 * h + 2 * vv;
    f[2 * vv]     = ar[kk];
    f[2 * vv + 1] = ar[kk + 1];
  }
  return f;
}
// A fragment where the logical A[M][K] is stored transposed as At[K][M]
__device__ __forceinline__ v16bf frag_at_bf16(const __bf16* At, int ldt, int row0,
                                              int k0, int l, int h) {
  v16bf f;
#pragma unroll
  for (int vv = 0; vv < 8; ++vv) {
    int kk = 16 * h + 2 * vv;
    f[2 * vv]     = At[(k0 + kk) * ldt + row0 + l];
    f[2 * vv + 1] = At[(k0 + kk + 1) * ldt + row0 + l];
  }
  return f;
}
// B 32x16 bf16: lane half h, N = lane&15.  VGPR v: K = 16h + 2v
__device__ __forceinline__ v16bf frag_b_bf16(const __bf16* Bm, int ldb, int k0,
                                             int col0, int l, int h) {
  v16bf f;
#pragma unroll
  for (int vv = 0; vv < 8; ++vv) {
    int kk = 16 * h + 2 * vv;
    f[2 * vv]     = Bm[(k0 + kk) * ldb + col0 + l];
    f[2 * vv + 1] = Bm[(k0 + kk + 1) * ldb + col0 + l];
  }
  return f;
}
// B fragment where logical B[K][N] is stored transposed as Bt[N][K]
__device__ __forceinline__ v16bf frag_bt_bf16(const __bf16* Bt, int ldt, int k0,
                                              int col0, int l, int h) {
  v16bf f;
  const __bf16* br = Bt + (col0 + l) * ldt + k0;
#pragma unroll
  for (int vv = 0; vv < 8; ++vv) {
    int kk = 16 * h + 2 * vv;
    f[2 * vv]     = br[kk];
    f[2 * vv + 1] = br[kk + 1];
  }
  return f;
}
// B fragment sourced from an f32 matrix (state S), converted to bf16 on read
__device__ __forceinline__ v16bf frag_b_f32cvt(const float* Bm, int ldb, int k0,
                                               int col0, int l, int h) {
  v16bf f;
#pragma unroll
  for (int vv = 0; vv < 8; ++vv) {
    int kk = 16 * h + 2 * vv;
    f[2 * vv]     = (__bf16)Bm[(k0 + kk) * ldb + col0 + l];
    f[2 * vv + 1] = (__bf16)Bm[(k0 + kk + 1) * ldb + col0 + l];
  }
  return f;
}

__device__ __forceinline__ v8f wmma_bf16(v16bf a, v16bf b, v8f c) {
  return __builtin_amdgcn_wmma_f32_16x16x32_bf16(false, a, false, b,
                                                 (short)0, c, false, false);
}

// ===========================================================================
// Kernel 1/3: phi = ratio * (exp(dd - diag - mx) + eps)  (bf16 output)
// dd[i][j] = sum_d (NRM*data[i][d]) * proj[j][d], GEMM via f32 16x16x4 WMMA.
// grid: BH_*64 blocks (64 rows each), 128 threads (4 waves).
// ===========================================================================
#define PHI_LDS_BYTES ((256 * 64 + 64 * 64 + 64 * 256) * 4)

__global__ __launch_bounds__(128) void FastAttention_phi_kernel(
    const float* __restrict__ data, const float* __restrict__ proj,
    __bf16* __restrict__ phi, const unsigned* __restrict__ keymax,
    int is_query) {
  extern __shared__ char smem[];
  float* projs = (float*)smem;            // [256][64]
  float* As    = projs + 256 * 64;        // [64][64]  scaled rows
  float* dd    = As + 64 * 64;            // [64][256]

  const int bh  = blockIdx.x >> 6;
  const int rt  = blockIdx.x & 63;        // 64-row tile within n
  const int tid = threadIdx.x;
  const int wave = tid >> 5, lane = tid & 31;
  const int l = lane & 15, h = lane >> 4;

  for (int i = tid; i < 256 * 64; i += 128) projs[i] = proj[i];
  const float* drow = data + ((size_t)bh * N_ + (size_t)rt * 64) * D_;
  for (int i = tid; i < 64 * 64; i += 128) As[i] = NRM_ * drow[i];
  __syncthreads();

#if __has_builtin(__builtin_amdgcn_wmma_f32_16x16x4_f32)
  // 4 row-tiles x 16 col-tiles of 16x16, K=64 (16 wmma steps each)
  for (int t = wave; t < 64; t += 4) {
    int ti = t >> 4, tj = t & 15;
    v8f acc = {};
#pragma unroll
    for (int k0 = 0; k0 < 64; k0 += 4) {
      v2f a, bb;
      a[0]  = As[(ti * 16 + l) * 64 + k0 + 2 * h];
      a[1]  = As[(ti * 16 + l) * 64 + k0 + 2 * h + 1];
      bb[0] = projs[(tj * 16 + l) * 64 + k0 + 2 * h];
      bb[1] = projs[(tj * 16 + l) * 64 + k0 + 2 * h + 1];
      acc = __builtin_amdgcn_wmma_f32_16x16x4_f32(false, a, false, bb,
                                                  (short)0, acc, false, false);
    }
#pragma unroll
    for (int r = 0; r < 8; ++r)
      dd[(ti * 16 + r + 8 * h) * 256 + tj * 16 + l] = acc[r];
  }
#else
  for (int idx = tid; idx < 64 * 256; idx += 128) {
    int i = idx >> 8, j = idx & 255;
    float s = 0.f;
    for (int d0 = 0; d0 < 64; ++d0) s += As[i * 64 + d0] * projs[j * 64 + d0];
    dd[idx] = s;
  }
#endif
  __syncthreads();

  if (tid < 64) {
    const float* arow = As + tid * 64;
    float diag = 0.f;
    for (int d0 = 0; d0 < 64; ++d0) diag += arow[d0] * arow[d0];
    diag *= 0.5f;                         // 0.5 * nrm^2 * sum(x^2)
    const float* ddr = dd + tid * 256;
    float mx;
    if (is_query) {
      mx = ddr[0];
      for (int j = 1; j < 256; ++j) mx = fmaxf(mx, ddr[j]);
    } else {
      mx = ordered_f32(keymax[bh]);
    }
    __bf16* prow = phi + ((size_t)bh * N_ + (size_t)rt * 64 + tid) * M_;
    for (int j = 0; j < 256; ++j)
      prow[j] = (__bf16)(RATIO_ * (__expf(ddr[j] - diag - mx) + EPSK_));
  }
}

// ===========================================================================
// Kernel 2: global max of dd over the whole [n x m] slice per (b,h) (keys)
// ===========================================================================
#define KMAX_LDS_BYTES ((256 * 64 + 64 * 64 + 128) * 4)

__global__ __launch_bounds__(128) void FastAttention_kmax_kernel(
    const float* __restrict__ data, const float* __restrict__ proj,
    unsigned* __restrict__ keymax) {
  extern __shared__ char smem[];
  float* projs = (float*)smem;
  float* As    = projs + 256 * 64;
  float* red   = As + 64 * 64;            // [128]

  const int bh  = blockIdx.x >> 6;
  const int rt  = blockIdx.x & 63;
  const int tid = threadIdx.x;
  const int wave = tid >> 5, lane = tid & 31;
  const int l = lane & 15, h = lane >> 4;

  for (int i = tid; i < 256 * 64; i += 128) projs[i] = proj[i];
  const float* drow = data + ((size_t)bh * N_ + (size_t)rt * 64) * D_;
  for (int i = tid; i < 64 * 64; i += 128) As[i] = NRM_ * drow[i];
  __syncthreads();

  float lmax = -3.4e38f;
#if __has_builtin(__builtin_amdgcn_wmma_f32_16x16x4_f32)
  for (int t = wave; t < 64; t += 4) {
    int ti = t >> 4, tj = t & 15;
    v8f acc = {};
#pragma unroll
    for (int k0 = 0; k0 < 64; k0 += 4) {
      v2f a, bb;
      a[0]  = As[(ti * 16 + l) * 64 + k0 + 2 * h];
      a[1]  = As[(ti * 16 + l) * 64 + k0 + 2 * h + 1];
      bb[0] = projs[(tj * 16 + l) * 64 + k0 + 2 * h];
      bb[1] = projs[(tj * 16 + l) * 64 + k0 + 2 * h + 1];
      acc = __builtin_amdgcn_wmma_f32_16x16x4_f32(false, a, false, bb,
                                                  (short)0, acc, false, false);
    }
#pragma unroll
    for (int r = 0; r < 8; ++r) lmax = fmaxf(lmax, acc[r]);
  }
#else
  for (int idx = tid; idx < 64 * 256; idx += 128) {
    int i = idx >> 8, j = idx & 255;
    float s = 0.f;
    for (int d0 = 0; d0 < 64; ++d0) s += As[i * 64 + d0] * projs[j * 64 + d0];
    lmax = fmaxf(lmax, s);
  }
#endif
  red[tid] = lmax;
  __syncthreads();
  if (tid == 0) {
    float m = red[0];
    for (int i = 1; i < 128; ++i) m = fmaxf(m, red[i]);
    atomicMax(&keymax[bh], f32_ordered(m));
  }
}

// ===========================================================================
// Kernel 4: chunked-recurrent causal linear attention.
// One block per (b,h); 512 threads = 16 waves; 32 sequential chunks.
// qb/kb staged via TDM (tensor_load_to_lds), gated by s_wait_tensorcnt.
// LDS: S f32[256][64], z f32[256], den f32[128],
//      qb/kb bf16[128][256], vb bf16[128][64], sc bf16[128][128]  = 247296 B
// ===========================================================================
#define ATTN_LDS_BYTES \
  ((256 * 64 + 256 + 128) * 4 + (128 * 256 * 2 + 128 * 64 + 128 * 128) * 2)

__global__ __launch_bounds__(512) void FastAttention_attn_kernel(
    const __bf16* __restrict__ phi_q, const __bf16* __restrict__ phi_k,
    const float* __restrict__ vsrc, float* __restrict__ out) {
  extern __shared__ char smem[];
  float*  S   = (float*)smem;                 // [256][64] running k v^T sum
  float*  z   = S + 256 * 64;                 // [256]     running k sum
  float*  den = z + 256;                      // [128]
  __bf16* qb  = (__bf16*)(den + 128);         // [128][256]
  __bf16* kb  = qb + 128 * 256;               // [128][256]
  __bf16* vb  = kb + 128 * 256;               // [128][64]
  __bf16* sc  = vb + 128 * 64;                // [128][128] masked scores

  const int bh   = blockIdx.x;
  const int tid  = threadIdx.x;
  const int wave = tid >> 5, lane = tid & 31;
  const int l = lane & 15, h = lane >> 4;

  for (int i = tid; i < 256 * 64; i += 512) S[i] = 0.f;
  for (int i = tid; i < 256; i += 512) z[i] = 0.f;
  __syncthreads();

  const __bf16* Qbh = phi_q + (size_t)bh * N_ * M_;
  const __bf16* Kbh = phi_k + (size_t)bh * N_ * M_;
  const float*  Vbh = vsrc + (size_t)bh * N_ * D_;
  float*        Obh = out + (size_t)bh * N_ * D_;

  for (int ch = 0; ch < NCHUNK; ++ch) {
    const __bf16* qs = Qbh + (size_t)ch * CHUNK_ * M_;
    const __bf16* ks = Kbh + (size_t)ch * CHUNK_ * M_;

    // ---- stage chunk: qb, kb via Tensor Data Mover; vb f32 -> bf16 -------
#if HAVE_TDM_
    if (tid < 32) {  // wave 0 issues both async tile copies (64KB each)
      tdm_copy_to_lds((unsigned)(uintptr_t)qb, qs, CHUNK_ * M_ * 2);
      tdm_copy_to_lds((unsigned)(uintptr_t)kb, ks, CHUNK_ * M_ * 2);
    }
#else
    {
      const unsigned* q32 = (const unsigned*)qs;
      const unsigned* k32 = (const unsigned*)ks;
      unsigned* qd = (unsigned*)qb;
      unsigned* kd = (unsigned*)kb;
      for (int i = tid; i < CHUNK_ * M_ / 2; i += 512) { qd[i] = q32[i]; kd[i] = k32[i]; }
    }
#endif
    const float* vs = Vbh + (size_t)ch * CHUNK_ * D_;
    for (int i = tid; i < CHUNK_ * D_; i += 512) vb[i] = (__bf16)vs[i];
    for (int i = tid; i < CHUNK_ * CHUNK_ / 2; i += 512) ((unsigned*)sc)[i] = 0u;

    // warm GL2 for the next chunk while this one computes
    if (ch + 1 < NCHUNK) {
      const char* qn = (const char*)(Qbh + (size_t)(ch + 1) * CHUNK_ * M_);
      const char* kn = (const char*)(Kbh + (size_t)(ch + 1) * CHUNK_ * M_);
      const char* vn = (const char*)(Vbh + (size_t)(ch + 1) * CHUNK_ * D_);
      __builtin_prefetch(qn + (size_t)tid * 128, 0, 1);  // 512*128B = 64KB
      __builtin_prefetch(kn + (size_t)tid * 128, 0, 1);
      __builtin_prefetch(vn + (size_t)tid * 64, 0, 1);   // 512*64B  = 32KB
    }
#if HAVE_TDM_
    __builtin_amdgcn_s_wait_tensorcnt(0);  // wave0's TDM done; others no-op
#endif
    __syncthreads();

    // ---- scores = tril(qb . kb^T): lower-triangular 16x16 tiles ----------
    for (int t = wave; t < 64; t += 16) {
      int ti = t >> 3, tj = t & 7;
      if (tj > ti) continue;              // upper tiles stay zero
      v8f acc = {};
#pragma unroll
      for (int k0 = 0; k0 < M_; k0 += 32) {
        v16bf a = frag_a_bf16(qb, M_, ti * 16, k0, l, h);
        v16bf b = frag_bt_bf16(kb, M_, k0, tj * 16, l, h);
        acc = wmma_bf16(a, b, acc);
      }
#pragma unroll
      for (int r = 0; r < 8; ++r) {
        int row = ti * 16 + r + 8 * h;
        int col = tj * 16 + l;
        sc[row * CHUNK_ + col] = (__bf16)((col <= row) ? acc[r] : 0.f);
      }
    }
    __syncthreads();

    // ---- denom[c] = q.(z+eps) + row-sum of masked scores -----------------
    // (sum_m q[c,m]*cumsum_{r<=c} k[r,m] == sum_{r<=c} scores[c,r])
    if (tid < CHUNK_) {
      float s = 0.f;
      const __bf16* qr = qb + tid * M_;
      for (int m = 0; m < M_; ++m) s += (float)qr[m] * (z[m] + EPSD_);
      const __bf16* sr = sc + tid * CHUNK_;
      for (int r = 0; r < CHUNK_; ++r) s += (float)sr[r];
      den[tid] = s;
    }
    __syncthreads();

    // ---- out = (qb.S + sc.vb) / denom  (8x4 output tiles) ----------------
    for (int t = wave; t < 32; t += 16) {
      int ti = t >> 2, tj = t & 3;
      v8f acc = {};
#pragma unroll
      for (int k0 = 0; k0 < M_; k0 += 32) {   // inter: K over m=256
        v16bf a = frag_a_bf16(qb, M_, ti * 16, k0, l, h);
        v16bf b = frag_b_f32cvt(S, D_, k0, tj * 16, l, h);
        acc = wmma_bf16(a, b, acc);
      }
#pragma unroll
      for (int k0 = 0; k0 < CHUNK_; k0 += 32) { // intra: K over chunk rows
        v16bf a = frag_a_bf16(sc, CHUNK_, ti * 16, k0, l, h);
        v16bf b = frag_b_bf16(vb, D_, k0, tj * 16, l, h);
        acc = wmma_bf16(a, b, acc);
      }
#pragma unroll
      for (int r = 0; r < 8; ++r) {
        int row = ti * 16 + r + 8 * h;
        int col = tj * 16 + l;
        Obh[((size_t)ch * CHUNK_ + row) * D_ + col] = acc[r] / den[row];
      }
    }
    __syncthreads();   // all reads of S (inter) done before S update

    // ---- S += kb^T . vb  (16x4 tiles of [256][64], f32 accumulate) -------
    for (int t = wave; t < 64; t += 16) {
      int ti = t >> 2, tj = t & 3;
      v8f acc;
#pragma unroll
      for (int r = 0; r < 8; ++r)
        acc[r] = S[(ti * 16 + r + 8 * h) * D_ + tj * 16 + l];
#pragma unroll
      for (int k0 = 0; k0 < CHUNK_; k0 += 32) {
        v16bf a = frag_at_bf16(kb, M_, ti * 16, k0, l, h);  // kb^T
        v16bf b = frag_b_bf16(vb, D_, k0, tj * 16, l, h);
        acc = wmma_bf16(a, b, acc);
      }
#pragma unroll
      for (int r = 0; r < 8; ++r)
        S[(ti * 16 + r + 8 * h) * D_ + tj * 16 + l] = acc[r];
    }
    // ---- z += column sums of kb ------------------------------------------
    if (tid < M_) {
      float s = 0.f;
      for (int c = 0; c < CHUNK_; ++c) s += (float)kb[c * M_ + tid];
      z[tid] += s;
    }
    __syncthreads();   // S/z updated before next chunk (and restaging)
  }
}

// ===========================================================================
extern "C" void kernel_launch(void* const* d_in, const int* in_sizes, int n_in,
                              void* d_out, int out_size, void* d_ws, size_t ws_size,
                              hipStream_t stream) {
  const float* q    = (const float*)d_in[0];
  const float* k    = (const float*)d_in[1];
  const float* v    = (const float*)d_in[2];
  const float* proj = (const float*)d_in[3];
  float* out = (float*)d_out;

  // workspace layout: phi_q bf16 | phi_k bf16 | keymax u32[16]
  const size_t phiElems = (size_t)BH_ * N_ * M_;          // 16,777,216
  char* ws = (char*)d_ws;
  __bf16*   phi_q  = (__bf16*)ws;
  __bf16*   phi_k  = (__bf16*)(ws + phiElems * sizeof(__bf16));
  unsigned* keymax = (unsigned*)(ws + 2 * phiElems * sizeof(__bf16));

  (void)hipFuncSetAttribute((const void*)FastAttention_phi_kernel,
                            hipFuncAttributeMaxDynamicSharedMemorySize,
                            PHI_LDS_BYTES);
  (void)hipFuncSetAttribute((const void*)FastAttention_kmax_kernel,
                            hipFuncAttributeMaxDynamicSharedMemorySize,
                            KMAX_LDS_BYTES);
  (void)hipFuncSetAttribute((const void*)FastAttention_attn_kernel,
                            hipFuncAttributeMaxDynamicSharedMemorySize,
                            ATTN_LDS_BYTES);

  // ordered-uint encoding: 0 decodes below every real float -> valid -inf init
  hipMemsetAsync(keymax, 0, BH_ * sizeof(unsigned), stream);

  dim3 phiGrid(BH_ * (N_ / 64));
  FastAttention_phi_kernel<<<phiGrid, 128, PHI_LDS_BYTES, stream>>>(
      q, proj, phi_q, keymax, /*is_query=*/1);
  FastAttention_kmax_kernel<<<phiGrid, 128, KMAX_LDS_BYTES, stream>>>(
      k, proj, keymax);
  FastAttention_phi_kernel<<<phiGrid, 128, PHI_LDS_BYTES, stream>>>(
      k, proj, phi_k, keymax, /*is_query=*/0);
  FastAttention_attn_kernel<<<dim3(BH_), 512, ATTN_LDS_BYTES, stream>>>(
      phi_q, phi_k, v, out);
}
